// CategoricalGATPolicy_17729624998135
// MI455X (gfx1250) — matrix operand order repaired and checked by reading, not compiled, not run on hardware
//
#include <hip/hip_runtime.h>
#include <hip/hip_bf16.h>
#include <math.h>

// Problem constants (from reference): N=50000, F=128, H=4, C=64, E=400000
#define GAT_F 128
#define GAT_H 4
#define GAT_C 64
#define NEG_SLOPE 0.2f

typedef __attribute__((ext_vector_type(16))) __bf16 v16bf;
typedef __attribute__((ext_vector_type(8)))  __bf16 v8bf;
typedef __attribute__((ext_vector_type(8)))  float  v8f;

// round-to-nearest-even f32 -> bf16
__device__ __forceinline__ __bf16 f2bf(float f) {
  union { float f; unsigned u; } x; x.f = f;
  unsigned r = x.u + 0x7FFFu + ((x.u >> 16) & 1u);
  union { unsigned short s; __bf16 b; } y; y.s = (unsigned short)(r >> 16);
  return y.b;
}

// elementwise f32 -> bf16 copy (4 per thread, n % 4 == 0)
__global__ void cvt_bf16_k(const float* __restrict__ in, __bf16* __restrict__ out, long n) {
  long i = (blockIdx.x * (long)blockDim.x + threadIdx.x) * 4;
  if (i >= n) return;
#pragma unroll
  for (int j = 0; j < 4; ++j) out[i + j] = f2bf(in[i + j]);
}

// Pack B[K,Nc] (f32, row-major) into WMMA fragment order, bf16:
// frag(tn, ks): lane l holds col tn*16+(l&15), K = ks*32 + ((l<16)?0:16) + i, i=0..15
// stored contiguously at Bp[(((tn*KS + ks)*32 + l)*16 + i]
__global__ void pack_B_k(const float* __restrict__ B, __bf16* __restrict__ Bp,
                         int K, int Nc) {
  int t = blockIdx.x * blockDim.x + threadIdx.x;
  int tilesN = Nc >> 4, ksteps = K >> 5;
  if (t >= tilesN * ksteps * 32) return;
  int lane = t & 31;
  int ks = (t >> 5) % ksteps;
  int tn = (t >> 5) / ksteps;
  int col = tn * 16 + (lane & 15);
  int kb  = ks * 32 + ((lane < 16) ? 0 : 16);
  __bf16* out = Bp + (long)t * 16;
#pragma unroll
  for (int i = 0; i < 16; ++i) out[i] = f2bf(B[(long)(kb + i) * Nc + col]);
}

// ---------------------------------------------------------------------------
// GEMM: Cout[M,Nc] = A[M,K] @ B[K,Nc] (+ Cin).
// A: bf16 row-major. Bp: bf16 pre-packed fragments. One wave per 16x64 strip
// of C (4 WMMA tiles) so the A fragment is reused 4x per k-step.
// M % 16 == 0, Nc % 64 == 0, K % 32 == 0 (all hold here).
// ---------------------------------------------------------------------------
template <int K>
__global__ void gemm_wmma(const __bf16* __restrict__ A, const __bf16* __restrict__ Bp,
                          const float* __restrict__ Cin, float* __restrict__ Cout,
                          int M, int Nc) {
  constexpr int KS = K / 32;
  int wid  = (int)((blockIdx.x * blockDim.x + threadIdx.x) >> 5);
  int lane = (int)(threadIdx.x & 31u);
  int groupsN = Nc >> 6;                       // 4 tiles (64 cols) per wave
  int nWaves  = (M >> 4) * groupsN;
  if (wid >= nWaves) return;                   // wave-uniform: EXEC all-ones
  int tm = wid / groupsN;
  int tg = wid - tm * groupsN;                 // tiles tg*4 .. tg*4+3
  int l15 = lane & 15;

  // A frag: lane<16 -> row l15, K {0..15}; lane>=16 -> K {16..31}
  const __bf16* Abase = A + (long)(tm * 16 + l15) * K + ((lane < 16) ? 0 : 16);
  const __bf16* Bbase = Bp + ((long)(tg * 4) * KS * 32 + lane) * 16;

  v8f c0 = {}, c1 = {}, c2 = {}, c3 = {};
#pragma unroll
  for (int ks = 0; ks < KS; ++ks) {
    v8bf alo = *(const v8bf*)(Abase + ks * 32);
    v8bf ahi = *(const v8bf*)(Abase + ks * 32 + 8);
    v16bf a;
#pragma unroll
    for (int i = 0; i < 8; ++i) { a[i] = alo[i]; a[i + 8] = ahi[i]; }
    const __bf16* bp = Bbase + (long)ks * 32 * 16;
    const long tstride = (long)KS * 32 * 16;   // next tn fragment
    v16bf b0 = *(const v16bf*)(bp);
    v16bf b1 = *(const v16bf*)(bp + tstride);
    v16bf b2 = *(const v16bf*)(bp + 2 * tstride);
    v16bf b3 = *(const v16bf*)(bp + 3 * tstride);
    c0 = __builtin_amdgcn_wmma_f32_16x16x32_bf16(false, a, false, b0, (short)0, c0, false, false);
    c1 = __builtin_amdgcn_wmma_f32_16x16x32_bf16(false, a, false, b1, (short)0, c1, false, false);
    c2 = __builtin_amdgcn_wmma_f32_16x16x32_bf16(false, a, false, b2, (short)0, c2, false, false);
    c3 = __builtin_amdgcn_wmma_f32_16x16x32_bf16(false, a, false, b3, (short)0, c3, false, false);
  }

  // C layout: VGPR j -> row (lane>=16?8:0)+j, col lane&15 (per 16x16 tile)
  int m0 = tm * 16 + ((lane >= 16) ? 8 : 0);
  v8f cc[4] = {c0, c1, c2, c3};
#pragma unroll
  for (int t = 0; t < 4; ++t) {
    int n = (tg * 4 + t) * 16 + l15;
#pragma unroll
    for (int j = 0; j < 8; ++j) {
      long idx = (long)(m0 + j) * Nc + n;
      float v = cc[t][j];
      if (Cin) v += Cin[idx];
      Cout[idx] = v;
    }
  }
}

// ---------------------------------------------------------------------------
__global__ void fill_f32(float* __restrict__ p, float v, long n) {
  long i = blockIdx.x * (long)blockDim.x + threadIdx.x;
  if (i < n) p[i] = v;
}

// a_src[t], a_dst[t] = <h[t,:], att_src/dst[head,:]>, t over N*Hh
__global__ void attn_scores(const float* __restrict__ h, const float* __restrict__ att_s,
                            const float* __restrict__ att_d, float* __restrict__ as,
                            float* __restrict__ ad, int NH, int Hh) {
  int t = blockIdx.x * blockDim.x + threadIdx.x;
  if (t >= NH) return;
  const float* hp = h + (long)t * GAT_C;
  const float* sp = att_s + (long)(t % Hh) * GAT_C;
  const float* dp = att_d + (long)(t % Hh) * GAT_C;
  float s = 0.f, d = 0.f;
#pragma unroll 8
  for (int c = 0; c < GAT_C; ++c) { float v = hp[c]; s += v * sp[c]; d += v * dp[c]; }
  as[t] = s;
  ad[t] = d;
}

// race-safe float atomic max via signed/unsigned int atomics
__device__ __forceinline__ void atomicMaxF(float* addr, float v) {
  if (v >= 0.f) atomicMax((int*)addr, __float_as_int(v));
  else          atomicMin((unsigned int*)addr, __float_as_uint(v));
}

__device__ __forceinline__ void edge_sd(const int* __restrict__ ei, long e, long E,
                                        int& s, int& d) {
  if (e < E) { s = ei[e]; d = ei[E + e]; }
  else       { s = d = (int)(e - E); }     // self loops appended after real edges
}

__global__ void edge_max_k(const int* __restrict__ ei, long E, long ET, int Hh,
                           const float* __restrict__ as, const float* __restrict__ ad,
                           float* __restrict__ m) {
  long t = blockIdx.x * (long)blockDim.x + threadIdx.x;
  if (t >= ET * Hh) return;
  long e = t / Hh; int hh = (int)(t - e * Hh);
  int s, d; edge_sd(ei, e, E, s, d);
  float ev = as[(long)s * Hh + hh] + ad[(long)d * Hh + hh];
  ev = (ev > 0.f) ? ev : NEG_SLOPE * ev;
  atomicMaxF(&m[(long)d * Hh + hh], ev);
}

__global__ void edge_expsum_k(const int* __restrict__ ei, long E, long ET, int Hh,
                              const float* __restrict__ as, const float* __restrict__ ad,
                              const float* __restrict__ m, float* __restrict__ denom) {
  long t = blockIdx.x * (long)blockDim.x + threadIdx.x;
  if (t >= ET * Hh) return;
  long e = t / Hh; int hh = (int)(t - e * Hh);
  int s, d; edge_sd(ei, e, E, s, d);
  long di = (long)d * Hh + hh;
  float ev = as[(long)s * Hh + hh] + ad[di];
  ev = (ev > 0.f) ? ev : NEG_SLOPE * ev;
  atomicAdd(&denom[di], expf(ev - m[di]));
}

__global__ void edge_msg_k(const int* __restrict__ ei, long E, long ET, int Hh,
                           const float* __restrict__ as, const float* __restrict__ ad,
                           const float* __restrict__ m, const float* __restrict__ denom,
                           const float* __restrict__ hbuf, float* __restrict__ acc,
                           float* __restrict__ alpha_out) {
  long t = blockIdx.x * (long)blockDim.x + threadIdx.x;
  if (t >= ET * Hh) return;
  long e = t / Hh; int hh = (int)(t - e * Hh);
  int s, d; edge_sd(ei, e, E, s, d);
  long di = (long)d * Hh + hh;
  float ev = as[(long)s * Hh + hh] + ad[di];
  ev = (ev > 0.f) ? ev : NEG_SLOPE * ev;
  float alpha = expf(ev - m[di]) / (denom[di] + 1e-16f);
  if (alpha_out) alpha_out[t] = alpha;     // only bound for layer 2 (Hh==1)
  const float* hp = hbuf + ((long)s * Hh + hh) * GAT_C;
  float* ap = acc + di * GAT_C;
#pragma unroll 4
  for (int c = 0; c < GAT_C; ++c) atomicAdd(&ap[c], hp[c] * alpha);
}

// p[i] += bias[col], then optional ELU
__global__ void bias_act_k(float* __restrict__ p, const float* __restrict__ bias,
                           int cols, long n, int do_elu) {
  long i = blockIdx.x * (long)blockDim.x + threadIdx.x;
  if (i >= n) return;
  float v = p[i] + bias[(int)(i % cols)];
  if (do_elu) v = (v > 0.f) ? v : expm1f(v);
  p[i] = v;
}

// L2-normalize rows of h2, write to d_out, compute logits = h2n @ w + b
__global__ void norm_logits_k(const float* __restrict__ h2, const float* __restrict__ w,
                              const float* __restrict__ b, float* __restrict__ h2o,
                              float* __restrict__ logits, int n) {
  int i = blockIdx.x * blockDim.x + threadIdx.x;
  if (i >= n) return;
  const float* r = h2 + (long)i * GAT_C;
  float ss = 0.f;
#pragma unroll 8
  for (int c = 0; c < GAT_C; ++c) ss += r[c] * r[c];
  float inv = 1.f / fmaxf(sqrtf(ss), 1e-12f);
  float lg = 0.f;
#pragma unroll 8
  for (int c = 0; c < GAT_C; ++c) {
    float v = r[c] * inv;
    h2o[(long)i * GAT_C + c] = v;
    lg += v * w[c];
  }
  logits[i] = lg + b[0];
}

__global__ void init_argmax_k(float* gmax, int* gidx) {
  if (threadIdx.x == 0 && blockIdx.x == 0) {
    *gmax = -__builtin_inff();
    *gidx = 0x7fffffff;
  }
}
__global__ void argmax_max_k(const float* __restrict__ logits, float* gmax, int n) {
  int i = blockIdx.x * blockDim.x + threadIdx.x;
  if (i < n) atomicMaxF(gmax, logits[i]);
}
__global__ void argmax_idx_k(const float* __restrict__ logits, const float* gmax,
                             int* gidx, int n) {
  int i = blockIdx.x * blockDim.x + threadIdx.x;
  if (i < n && logits[i] == *gmax) atomicMin(gidx, i);   // first max index
}
__global__ void write_action_k(const int* gidx, float* out) {
  if (threadIdx.x == 0 && blockIdx.x == 0) out[0] = (float)(*gidx);
}

// ---------------------------------------------------------------------------
extern "C" void kernel_launch(void* const* d_in, const int* in_sizes, int n_in,
                              void* d_out, int out_size, void* d_ws, size_t ws_size,
                              hipStream_t stream) {
  const float* x    = (const float*)d_in[0];
  const int*   ei   = (const int*)  d_in[1];
  const float* W1   = (const float*)d_in[2];
  const float* as1w = (const float*)d_in[3];
  const float* ad1w = (const float*)d_in[4];
  const float* b1   = (const float*)d_in[5];
  const float* W2   = (const float*)d_in[6];
  const float* as2w = (const float*)d_in[7];
  const float* ad2w = (const float*)d_in[8];
  const float* b2   = (const float*)d_in[9];
  const float* Wr1  = (const float*)d_in[10];
  const float* Wr2  = (const float*)d_in[11];
  const float* wact = (const float*)d_in[12];
  const float* bact = (const float*)d_in[13];

  const int  N  = in_sizes[0] / GAT_F;   // 50000 (divisible by 16)
  const long E  = in_sizes[1] / 2;       // 400000
  const long ET = E + N;                 // with self loops
  const int  HC = GAT_H * GAT_C;         // 256

  // ---- workspace carve-out: f32 region, then 32B-aligned bf16 region ----
  float* ws   = (float*)d_ws;
  float* h1   = ws;  ws += (long)N * HC;     // x@W1 (pre-attention features)
  float* acc1 = ws;  ws += (long)N * HC;     // message sums -> h1 final
  float* sa1  = ws;  ws += (long)N * GAT_H;
  float* da1  = ws;  ws += (long)N * GAT_H;
  float* m1   = ws;  ws += (long)N * GAT_H;
  float* dn1  = ws;  ws += (long)N * GAT_H;
  float* h2b  = ws;  ws += (long)N * GAT_C;  // h1@W2
  float* acc2 = ws;  ws += (long)N * GAT_C;
  float* sa2  = ws;  ws += N;
  float* da2  = ws;  ws += N;
  float* m2   = ws;  ws += N;
  float* dn2  = ws;  ws += N;
  float* gmax = ws;  ws += 1;
  int*   gidx = (int*)ws; ws += 1;

  __bf16* bfp  = (__bf16*)(((uintptr_t)ws + 31) & ~(uintptr_t)31);
  __bf16* xbf  = bfp;  bfp += (long)N * GAT_F;   // x in bf16
  __bf16* h1bf = bfp;  bfp += (long)N * HC;      // h1 (post layer1) in bf16
  __bf16* W1p  = bfp;  bfp += (long)GAT_F * HC;
  __bf16* Wr1p = bfp;  bfp += (long)GAT_F * HC;
  __bf16* W2p  = bfp;  bfp += (long)HC * GAT_C;
  __bf16* Wr2p = bfp;  bfp += (long)HC * GAT_C;

  // output layout: [action(1)] [h2 N*64] [alpha2 ET] [logits N]
  float* out_f     = (float*)d_out;
  float* h2_out    = out_f + 1;
  float* alpha_out = h2_out + (long)N * GAT_C;
  float* lg_out    = alpha_out + ET;

  auto nb = [](long n, int bs) { return (unsigned)((n + bs - 1) / bs); };

  // ---- precompute bf16 operands ----
  cvt_bf16_k<<<nb((long)N * GAT_F / 4, 256), 256, 0, stream>>>(x, xbf, (long)N * GAT_F);
  pack_B_k<<<nb((HC / 16) * (GAT_F / 32) * 32, 256), 256, 0, stream>>>(W1,  W1p,  GAT_F, HC);
  pack_B_k<<<nb((HC / 16) * (GAT_F / 32) * 32, 256), 256, 0, stream>>>(Wr1, Wr1p, GAT_F, HC);
  pack_B_k<<<nb((GAT_C / 16) * (HC / 32) * 32, 256), 256, 0, stream>>>(W2,  W2p,  HC, GAT_C);
  pack_B_k<<<nb((GAT_C / 16) * (HC / 32) * 32, 256), 256, 0, stream>>>(Wr2, Wr2p, HC, GAT_C);

  // ---- layer 1 ----
  {
    long waves = (long)(N / 16) * (HC / 64);
    gemm_wmma<GAT_F><<<nb(waves * 32, 256), 256, 0, stream>>>(xbf, W1p, nullptr, h1, N, HC);
  }
  attn_scores<<<nb((long)N * GAT_H, 256), 256, 0, stream>>>(h1, as1w, ad1w, sa1, da1,
                                                            N * GAT_H, GAT_H);
  fill_f32<<<nb((long)N * GAT_H, 256), 256, 0, stream>>>(m1, -INFINITY, (long)N * GAT_H);
  fill_f32<<<nb((long)N * GAT_H, 256), 256, 0, stream>>>(dn1, 0.f, (long)N * GAT_H);
  fill_f32<<<nb((long)N * HC, 256), 256, 0, stream>>>(acc1, 0.f, (long)N * HC);
  edge_max_k   <<<nb(ET * GAT_H, 256), 256, 0, stream>>>(ei, E, ET, GAT_H, sa1, da1, m1);
  edge_expsum_k<<<nb(ET * GAT_H, 256), 256, 0, stream>>>(ei, E, ET, GAT_H, sa1, da1, m1, dn1);
  edge_msg_k   <<<nb(ET * GAT_H, 256), 256, 0, stream>>>(ei, E, ET, GAT_H, sa1, da1, m1, dn1,
                                                         h1, acc1, nullptr);
  bias_act_k<<<nb((long)N * HC, 256), 256, 0, stream>>>(acc1, b1, HC, (long)N * HC, 1);
  {
    long waves = (long)(N / 16) * (HC / 64);
    gemm_wmma<GAT_F><<<nb(waves * 32, 256), 256, 0, stream>>>(xbf, Wr1p, acc1, acc1, N, HC);
  }

  // ---- layer 2 ----
  cvt_bf16_k<<<nb((long)N * HC / 4, 256), 256, 0, stream>>>(acc1, h1bf, (long)N * HC);
  {
    long waves = (long)(N / 16) * (GAT_C / 64);
    gemm_wmma<256><<<nb(waves * 32, 256), 256, 0, stream>>>(h1bf, W2p, nullptr, h2b, N, GAT_C);
  }
  attn_scores<<<nb(N, 256), 256, 0, stream>>>(h2b, as2w, ad2w, sa2, da2, N, 1);
  fill_f32<<<nb(N, 256), 256, 0, stream>>>(m2, -INFINITY, (long)N);
  fill_f32<<<nb(N, 256), 256, 0, stream>>>(dn2, 0.f, (long)N);
  fill_f32<<<nb((long)N * GAT_C, 256), 256, 0, stream>>>(acc2, 0.f, (long)N * GAT_C);
  edge_max_k   <<<nb(ET, 256), 256, 0, stream>>>(ei, E, ET, 1, sa2, da2, m2);
  edge_expsum_k<<<nb(ET, 256), 256, 0, stream>>>(ei, E, ET, 1, sa2, da2, m2, dn2);
  edge_msg_k   <<<nb(ET, 256), 256, 0, stream>>>(ei, E, ET, 1, sa2, da2, m2, dn2,
                                                 h2b, acc2, alpha_out);
  bias_act_k<<<nb((long)N * GAT_C, 256), 256, 0, stream>>>(acc2, b2, GAT_C,
                                                           (long)N * GAT_C, 0);
  {
    long waves = (long)(N / 16) * (GAT_C / 64);
    gemm_wmma<256><<<nb(waves * 32, 256), 256, 0, stream>>>(h1bf, Wr2p, acc2, acc2, N, GAT_C);
  }

  // ---- normalize, logits, argmax ----
  norm_logits_k<<<nb(N, 256), 256, 0, stream>>>(acc2, wact, bact, h2_out, lg_out, N);
  init_argmax_k<<<1, 1, 0, stream>>>(gmax, gidx);
  argmax_max_k<<<nb(N, 256), 256, 0, stream>>>(lg_out, gmax, N);
  argmax_idx_k<<<nb(N, 256), 256, 0, stream>>>(lg_out, gmax, gidx, N);
  write_action_k<<<1, 1, 0, stream>>>(gidx, out_f);
}
